// MultiHeadAttention_9483287790196
// MI455X (gfx1250) — compile-verified, matrix operand order checked by reading, not run on hardware
//
#include <hip/hip_runtime.h>

#define EMB   128
#define HID   128
#define NH    16
#define MODEL 2048
#define BB    4
#define SS    2048
#define WPB   4   // waves per block (wave32)

typedef _Float16 half_t;
typedef __attribute__((ext_vector_type(16))) _Float16 v16h;
typedef __attribute__((ext_vector_type(8)))  _Float16 v8h;
typedef __attribute__((ext_vector_type(8)))  float    v8f;
typedef unsigned int u32x4 __attribute__((ext_vector_type(4)));
typedef int          i32x4 __attribute__((ext_vector_type(4)));
typedef int          i32x8 __attribute__((ext_vector_type(8)));

static __device__ __forceinline__ v8f wmma16(v16h a, v16h b, v8f c) {
  // D = A(16x32 f16) * B(32x16 f16) + C(16x16 f32)
  return __builtin_amdgcn_wmma_f32_16x16x32_f16(false, a, false, b, (short)0, c,
                                                false, false);
}

// A/B fragment from row-major f16 (16 rows x 32 "k"), ld in halves.
// Lane L (wave32): row = L&15, k-half base = (L>>4)*8 ; halves k..k+7, k+16..k+23.
static __device__ __forceinline__ v16h frag_f16(const half_t* __restrict__ p, int ld) {
  int lane = threadIdx.x & 31;
  const half_t* q = p + (size_t)(lane & 15) * ld + ((lane >> 4) << 3);
  v8h lo = *(const v8h*)q;
  v8h hi = *(const v8h*)(q + 16);
  return __builtin_shufflevector(lo, hi, 0,1,2,3,4,5,6,7,8,9,10,11,12,13,14,15);
}

// Same fragment but converting from fp32 source on the fly.
static __device__ __forceinline__ v16h frag_f32(const float* __restrict__ p, int ld) {
  int lane = threadIdx.x & 31;
  const float* q = p + (size_t)(lane & 15) * ld + ((lane >> 4) << 3);
  v16h r;
#pragma unroll
  for (int i = 0; i < 8; ++i) { r[i] = (half_t)q[i]; r[i + 8] = (half_t)q[16 + i]; }
  return r;
}

// Pre-packed B fragment: contiguous 32B per lane.
static __device__ __forceinline__ v16h frag_packed(const half_t* __restrict__ pk,
                                                   int kstep, int nt, int NT) {
  int lane = threadIdx.x & 31;
  return *(const v16h*)(pk + (((size_t)kstep * NT + nt) << 9) + ((size_t)lane << 4));
}

// Tensor Data Mover: 2D tile load Global -> LDS, 2-byte elements.
// Descriptor layout per CDNA5 ISA ch.8 (D# group0 / group1); groups 2/3 zero (2D).
static __device__ __forceinline__ void tdm_load_2d(unsigned lds_addr,
                                                   const half_t* gptr,
                                                   unsigned dim0, unsigned dim1,
                                                   unsigned tile0, unsigned tile1,
                                                   unsigned stride0) {
  unsigned long long ga = (unsigned long long)(size_t)gptr;
  u32x4 g0;
  g0[0] = 1u;                                       // count=1, user descriptor
  g0[1] = lds_addr;                                 // LDS byte address
  g0[2] = (unsigned)(ga & 0xffffffffu);             // global_addr[31:0]
  g0[3] = (unsigned)((ga >> 32) & 0x01ffffffu) | 0x80000000u; // addr[56:32], type=2
  i32x8 g1;
  g1[0] = 0x00010000;                               // data_size=1 (2B), no multicast
  g1[1] = (int)((dim0 & 0xffffu) << 16);            // tensor_dim0[15:0] @ [63:48]
  g1[2] = (int)(((dim0 >> 16) & 0xffffu) | ((dim1 & 0xffffu) << 16));
  g1[3] = (int)(((dim1 >> 16) & 0xffffu) | ((tile0 & 0xffffu) << 16)); // tile_dim0
  g1[4] = (int)(tile1 & 0xffffu);                   // tile_dim1 (tile_dim2 = 0)
  g1[5] = (int)stride0;                             // tensor_dim0_stride[31:0]
  g1[6] = 0;                                        // stride0[47:32], stride1[15:0]
  g1[7] = 0;
  i32x4 z4 = (i32x4)0;
#if defined(__clang_major__) && (__clang_major__ >= 23)
  i32x8 z8 = (i32x8)0;
  __builtin_amdgcn_tensor_load_to_lds(g0, g1, z4, z4, z8, 0);
#else
  __builtin_amdgcn_tensor_load_to_lds(g0, g1, z4, z4, 0);
#endif
}

// Shuffle a row-major (K,N) fp32 weight into f16 B-operand fragment order.
__global__ void pack_w_kernel(const float* __restrict__ W, half_t* __restrict__ packed,
                              int K, int N) {
  int idx = blockIdx.x * blockDim.x + threadIdx.x;
  int total = K * N;
  if (idx >= total) return;
  int NT = N >> 4;
  int f = idx >> 9;            // fragment = kstep*NT + nt (512 halves each)
  int within = idx & 511;
  int lane = within >> 4;
  int e = within & 15;         // element within v16h
  int kstep = f / NT;
  int nt = f - kstep * NT;
  int n = (nt << 4) + (lane & 15);
  int khalf = (lane >> 4) << 3;
  int j = e >> 1, hbit = e & 1;
  int k = (kstep << 5) +
          ((j < 4) ? (khalf + 2 * j + hbit) : (16 + khalf + 2 * (j - 4) + hbit));
  packed[idx] = (half_t)W[(size_t)k * N + n];
}

// QKV projection: one wave -> 16 seq rows x 128 hidden for one head.
__global__ void __launch_bounds__(WPB * 32)
proj_kernel(const float* __restrict__ qin, const float* __restrict__ kin,
            const float* __restrict__ vin,
            const half_t* __restrict__ wqp, const half_t* __restrict__ wkp,
            const half_t* __restrict__ wvp,
            const float* __restrict__ bq, const float* __restrict__ bk,
            const float* __restrict__ bv,
            half_t* __restrict__ qp, half_t* __restrict__ kp,
            half_t* __restrict__ vT) {
  int wave = threadIdx.x >> 5;
  int lane = threadIdx.x & 31;
  int gid = blockIdx.x * WPB + wave;       // (tile, head)
  int h = gid % NH;
  int tile = gid / NH;
  int b = tile / (SS / 16);
  int s0 = (tile % (SS / 16)) << 4;
  int which = blockIdx.y;                  // 0=q 1=k 2=v

  const float*  x    = (which == 0) ? qin : (which == 1) ? kin : vin;
  const half_t* wp   = (which == 0) ? wqp : (which == 1) ? wkp : wvp;
  const float*  bias = (which == 0) ? bq  : (which == 1) ? bk  : bv;

  const float* xa = x + ((size_t)b * SS + s0) * MODEL + h * EMB;
  v8f acc[8];
#pragma unroll
  for (int i = 0; i < 8; ++i) acc[i] = (v8f)0.f;

#pragma unroll
  for (int kk = 0; kk < 4; ++kk) {          // K = 128 = 4 x 32
    v16h a = frag_f32(xa + kk * 32, MODEL);
#pragma unroll
    for (int nt = 0; nt < 8; ++nt) {        // N = 128 = 8 x 16
      v16h bf = frag_packed(wp, kk, nt, 8);
      acc[nt] = wmma16(a, bf, acc[nt]);
    }
  }

  int rrow = (lane >> 4) << 3;              // C layout: M = r + 8*(lane/16)
  int col0 = lane & 15;                     // N = lane&15
  if (which < 2) {
    half_t* base = ((which == 0) ? qp : kp) + ((size_t)(b * NH + h) * SS + s0) * HID;
#pragma unroll
    for (int nt = 0; nt < 8; ++nt) {
      int col = nt * 16 + col0;
      float bb_ = bias[col];
#pragma unroll
      for (int r = 0; r < 8; ++r)
        base[(size_t)(rrow + r) * HID + col] = (half_t)(acc[nt][r] + bb_);
    }
  } else {
    // store V transposed: vT[b,h,d,key] so P*V B-operand loads are contiguous
    half_t* base = vT + (size_t)(b * NH + h) * HID * SS + s0;
#pragma unroll
    for (int nt = 0; nt < 8; ++nt) {
      int col = nt * 16 + col0;
      float bb_ = bias[col];
#pragma unroll
      for (int r = 0; r < 8; ++r)
        base[(size_t)col * SS + (rrow + r)] = (half_t)(acc[nt][r] + bb_);
    }
  }
}

// Flash attention. One block (4 waves) = one (b, h, 64-query group); waves each
// own a 16-query tile. K/V tiles (32 keys) staged into LDS by the Tensor Data
// Mover, double-buffered: DMA of tile t+1 overlaps WMMA compute of tile t.
__global__ void __launch_bounds__(WPB * 32)
attn_kernel(const half_t* __restrict__ qp, const half_t* __restrict__ kp,
            const half_t* __restrict__ vT, half_t* __restrict__ ctx) {
  __shared__ __align__(16) half_t k_buf[2][32 * HID];  // 2 x 8 KB: 32 keys x 128
  __shared__ __align__(16) half_t v_buf[2][HID * 32];  // 2 x 8 KB: 128 d x 32 keys
  __shared__ __align__(16) half_t p_buf[WPB][16 * 32]; // per-wave P transpose

  int wave = threadIdx.x >> 5;
  int lane = threadIdx.x & 31;
  int qg = blockIdx.x % (SS / 64);          // 64-query group
  int bh = blockIdx.x / (SS / 64);
  int h  = bh % NH;
  int b  = bh / NH;
  int qt = qg * WPB + wave;                 // this wave's 16-query tile

  const half_t* qph = qp + (size_t)(b * NH + h) * SS * HID;
  const half_t* kph = kp + (size_t)(b * NH + h) * SS * HID;
  const half_t* vth = vT + (size_t)(b * NH + h) * HID * SS;
  half_t* lp = p_buf[wave];

  v16h qf[4];
#pragma unroll
  for (int kk = 0; kk < 4; ++kk)
    qf[kk] = frag_f16(qph + (size_t)(qt * 16) * HID + kk * 32, HID);

  float m[8], l[8];
#pragma unroll
  for (int r = 0; r < 8; ++r) { m[r] = -1e30f; l[r] = 0.f; }
  v8f o[8];
#pragma unroll
  for (int i = 0; i < 8; ++i) o[i] = (v8f)0.f;

  const float sc = 0.088388347648318447f;   // 1/sqrt(128)
  int rrow = (lane >> 4) << 3;
  int col0 = lane & 15;
  const int NKT = SS / 32;                  // 64 key tiles

  // prologue: TDM-stage tile 0 into buffer 0 (wave 0 issues, EXEC-independent)
  if (wave == 0) {
    tdm_load_2d((unsigned)(size_t)&k_buf[0][0], kph, HID, 32, HID, 32, HID);
    tdm_load_2d((unsigned)(size_t)&v_buf[0][0], vth, 32, HID, 32, HID, SS);
    __builtin_amdgcn_s_wait_tensorcnt(0);
  }
  __syncthreads();

  for (int kt = 0; kt < NKT; ++kt) {
    int par = kt & 1;
    if (wave == 0 && kt + 1 < NKT) {        // async-stage next tile
      int key1 = (kt + 1) * 32;
      tdm_load_2d((unsigned)(size_t)&k_buf[par ^ 1][0],
                  kph + (size_t)key1 * HID, HID, 32, HID, 32, HID);
      tdm_load_2d((unsigned)(size_t)&v_buf[par ^ 1][0],
                  vth + key1, 32, HID, 32, HID, SS);
    }
    const half_t* kb = k_buf[par];
    const half_t* vb = v_buf[par];

    v8f s0v = (v8f)0.f, s1v = (v8f)0.f;
#pragma unroll
    for (int kk = 0; kk < 4; ++kk) {        // scores 16x32 over hidden=128
      v16h b0 = frag_f16(kb + kk * 32, HID);
      v16h b1 = frag_f16(kb + 16 * HID + kk * 32, HID);
      s0v = wmma16(qf[kk], b0, s0v);
      s1v = wmma16(qf[kk], b1, s1v);
    }

    float alpha[8];
#pragma unroll
    for (int r = 0; r < 8; ++r) {
      float a0 = s0v[r] * sc;
      float a1 = s1v[r] * sc;
      float rmax = fmaxf(a0, a1);
#pragma unroll
      for (int mask = 1; mask < 16; mask <<= 1)       // row reduce in 16-lane half
        rmax = fmaxf(rmax, __shfl_xor(rmax, mask, 32));
      float mn = fmaxf(m[r], rmax);
      float p0 = __expf(a0 - mn);
      float p1 = __expf(a1 - mn);
      float rs = p0 + p1;
#pragma unroll
      for (int mask = 1; mask < 16; mask <<= 1)
        rs += __shfl_xor(rs, mask, 32);
      alpha[r] = __expf(m[r] - mn);
      l[r] = l[r] * alpha[r] + rs;
      m[r] = mn;
      // C-layout -> row-major 16x32 P tile in LDS (f16)
      lp[(rrow + r) * 32 + col0]      = (half_t)p0;
      lp[(rrow + r) * 32 + 16 + col0] = (half_t)p1;
    }
#pragma unroll
    for (int nt = 0; nt < 8; ++nt)
#pragma unroll
      for (int r = 0; r < 8; ++r)
        o[nt][r] *= alpha[r];

    asm volatile("s_wait_dscnt 0x0" ::: "memory");
    v16h pf = frag_f16(lp, 32);             // A-operand layout via LDS transpose
#pragma unroll
    for (int nt = 0; nt < 8; ++nt) {        // O += P(16x32) * V(32x128)
      v16h vf = frag_f16(vb + (size_t)(nt * 16) * 32, 32);
      o[nt] = wmma16(pf, vf, o[nt]);
    }

    if (wave == 0 && kt + 1 < NKT)
      __builtin_amdgcn_s_wait_tensorcnt(0); // next tile resident before barrier
    __syncthreads();                        // also fences buffer reuse
  }

  float inv[8];
#pragma unroll
  for (int r = 0; r < 8; ++r) inv[r] = 1.f / l[r];
  half_t* cbase = ctx + ((size_t)b * SS + qt * 16) * MODEL + h * HID;
#pragma unroll
  for (int nt = 0; nt < 8; ++nt) {
    int col = nt * 16 + col0;
#pragma unroll
    for (int r = 0; r < 8; ++r)
      cbase[(size_t)(rrow + r) * MODEL + col] = (half_t)(o[nt][r] * inv[r]);
  }
}

// Output projection: (B*S, 2048) x (2048, 2048) -> fp32 out (+bo).
__global__ void __launch_bounds__(WPB * 32)
out_gemm_kernel(const half_t* __restrict__ ctx, const half_t* __restrict__ wop,
                const float* __restrict__ bo, float* __restrict__ out) {
  int wave = threadIdx.x >> 5;
  int lane = threadIdx.x & 31;
  int gid = blockIdx.x * WPB + wave;
  const int NGRP = MODEL / 64;              // 32 groups of 4 n-tiles
  int mt = gid / NGRP;
  int ng = gid % NGRP;
  int row0 = mt * 16;
  v8f acc[4];
#pragma unroll
  for (int i = 0; i < 4; ++i) acc[i] = (v8f)0.f;
  const half_t* abase = ctx + (size_t)row0 * MODEL;
  for (int kstep = 0; kstep < MODEL / 32; ++kstep) {
    v16h a = frag_f16(abase + kstep * 32, MODEL);
#pragma unroll
    for (int j = 0; j < 4; ++j) {
      v16h bf = frag_packed(wop, kstep, ng * 4 + j, MODEL / 16);
      acc[j] = wmma16(a, bf, acc[j]);
    }
  }
  int rrow = (lane >> 4) << 3;
  int col0 = lane & 15;
#pragma unroll
  for (int j = 0; j < 4; ++j) {
    int col = ng * 64 + j * 16 + col0;
    float bb_ = bo[col];
#pragma unroll
    for (int r = 0; r < 8; ++r)
      out[(size_t)(row0 + rrow + r) * MODEL + col] = acc[j][r] + bb_;
  }
}

extern "C" void kernel_launch(void* const* d_in, const int* in_sizes, int n_in,
                              void* d_out, int out_size, void* d_ws, size_t ws_size,
                              hipStream_t stream) {
  (void)in_sizes; (void)n_in; (void)out_size; (void)ws_size;
  const float* q  = (const float*)d_in[0];
  const float* k  = (const float*)d_in[1];
  const float* v  = (const float*)d_in[2];
  const float* Wq = (const float*)d_in[3];
  const float* bq = (const float*)d_in[4];
  const float* Wk = (const float*)d_in[5];
  const float* bk = (const float*)d_in[6];
  const float* Wv = (const float*)d_in[7];
  const float* bv = (const float*)d_in[8];
  const float* Wo = (const float*)d_in[9];
  const float* bo = (const float*)d_in[10];
  float* out = (float*)d_out;

  char* ws = (char*)d_ws;
  const size_t SZP = (size_t)BB * NH * SS * HID * sizeof(half_t); // 32 MiB each
  half_t* qp  = (half_t*)(ws);
  half_t* kp  = (half_t*)(ws + SZP);
  half_t* vT  = (half_t*)(ws + 2 * SZP);
  half_t* ctx = (half_t*)(ws + 3 * SZP);
  half_t* wqp = (half_t*)(ws + 4 * SZP);
  half_t* wkp = wqp + EMB * HID;
  half_t* wvp = wkp + EMB * HID;
  half_t* wop = wvp + EMB * HID;            // 2048*2048 f16

  // 1) pack weights into WMMA B-fragment order (f32 -> f16)
  pack_w_kernel<<<(EMB * HID + 255) / 256, 256, 0, stream>>>(Wq, wqp, EMB, HID);
  pack_w_kernel<<<(EMB * HID + 255) / 256, 256, 0, stream>>>(Wk, wkp, EMB, HID);
  pack_w_kernel<<<(EMB * HID + 255) / 256, 256, 0, stream>>>(Wv, wvp, EMB, HID);
  pack_w_kernel<<<(MODEL * MODEL + 255) / 256, 256, 0, stream>>>(Wo, wop, MODEL, MODEL);

  // 2) QKV projections (V stored transposed)
  dim3 pgrid(BB * (SS / 16) * NH / WPB, 3);
  proj_kernel<<<pgrid, WPB * 32, 0, stream>>>(q, k, v, wqp, wkp, wvp,
                                              bq, bk, bv, qp, kp, vT);

  // 3) flash attention with TDM-staged K/V tiles
  attn_kernel<<<BB * NH * (SS / 64), WPB * 32, 0, stream>>>(qp, kp, vT, ctx);

  // 4) output GEMM + bias
  out_gemm_kernel<<<(BB * SS / 16) * (MODEL / 64) / WPB, WPB * 32, 0, stream>>>(
      ctx, wop, bo, out);
}
